// embE3Conv_87136296501551
// MI455X (gfx1250) — compile-verified, matrix operand order checked by reading, not run on hardware
//
#include <hip/hip_runtime.h>
#include <math.h>

typedef __attribute__((ext_vector_type(16))) _Float16 v16h;
typedef __attribute__((ext_vector_type(8)))  _Float16 v8h;
typedef __attribute__((ext_vector_type(8)))  float    v8f;

#define WPB 4          // waves per block in edge kernel
#define NPATH 13
#define NFRAG 94       // 8 (w1) + 8 (w2) + 78 (w3) fragments
#define W3J_FLOATS 573

// Fast transcendentals: native v_exp_f32 (exp2) and v_rcp_f32.
__device__ inline float fast_exp(float x) {
  return __builtin_amdgcn_exp2f(x * 1.4426950408889634f);
}
__device__ inline float fast_sigmoid(float x) {
  return __builtin_amdgcn_rcpf(1.0f + fast_exp(-x));
}

// ---------------------------------------------------------------------------
// Init kernel: Wigner-3j tables (double precision, ported from reference) and
// the silu normalization constant, written into workspace.
// ---------------------------------------------------------------------------
struct cdbl { double re, im; };
__device__ inline cdbl cmul(cdbl a, cdbl b) {
  return {a.re * b.re - a.im * b.im, a.re * b.im + a.im * b.re};
}
__device__ inline double dfact(int n) {
  double r = 1.0;
  for (int i = 2; i <= n; ++i) r *= (double)i;
  return r;
}

__device__ void su2_cg(int j1, int j2, int j3, double C[7][5][5]) {
  for (int a = 0; a < 7; a++)
    for (int b = 0; b < 5; b++)
      for (int c = 0; c < 5; c++) C[a][b][c] = 0.0;
  for (int m1 = -j1; m1 <= j1; m1++)
    for (int m2 = -j2; m2 <= j2; m2++) {
      int M = m1 + m2;
      if (M > j3 || M < -j3) continue;
      double pref = sqrt((2.0 * j3 + 1.0) * dfact(j3 + j1 - j2) * dfact(j3 - j1 + j2) *
                         dfact(j1 + j2 - j3) / dfact(j1 + j2 + j3 + 1));
      pref *= sqrt(dfact(j3 + M) * dfact(j3 - M) * dfact(j1 - m1) * dfact(j1 + m1) *
                   dfact(j2 - m2) * dfact(j2 + m2));
      double s = 0.0;
      for (int k = 0; k <= j1 + j2 - j3; k++) {
        int d0 = k, d1 = j1 + j2 - j3 - k, d2 = j1 - m1 - k, d3 = j2 + m2 - k;
        int d4 = j3 - j2 + m1 + k, d5 = j3 - j1 - m2 + k;
        if (d0 < 0 || d1 < 0 || d2 < 0 || d3 < 0 || d4 < 0 || d5 < 0) continue;
        double denom = dfact(d0) * dfact(d1) * dfact(d2) * dfact(d3) * dfact(d4) * dfact(d5);
        s += ((k & 1) ? -1.0 : 1.0) / denom;
      }
      C[j1 + m1][j2 + m2][j3 + M] = pref * s;
    }
}

__device__ void qmat(int l, cdbl Q[7][7]) {
  for (int a = 0; a < 7; a++)
    for (int b = 0; b < 7; b++) Q[a][b] = {0.0, 0.0};
  double is2 = 1.0 / sqrt(2.0);
  for (int m = -l; m < 0; m++) {
    Q[l + m][l - m] = {is2, 0.0};
    Q[l + m][l + m] = {0.0, -is2};
  }
  Q[l][l] = {1.0, 0.0};
  for (int m = 1; m <= l; m++) {
    double sgn = (m & 1) ? -1.0 : 1.0;
    Q[l + m][l + m] = {sgn * is2, 0.0};
    Q[l + m][l - m] = {0.0, sgn * is2};
  }
  cdbl ph;  // (-i)^l
  int lm = l & 3;
  if (lm == 0)      ph = { 1.0,  0.0};
  else if (lm == 1) ph = { 0.0, -1.0};
  else if (lm == 2) ph = {-1.0,  0.0};
  else              ph = { 0.0,  1.0};
  for (int a = 0; a < 7; a++)
    for (int b = 0; b < 7; b++) Q[a][b] = cmul(ph, Q[a][b]);
}

__global__ void init_tables(float* __restrict__ w3j, float* __restrict__ cst) {
  __shared__ double red[256];
  const int tid = threadIdx.x;

  // silu normalization constant (200001-point quadrature, matches reference)
  double part = 0.0;
  for (int i = tid; i < 200001; i += 256) {
    double z = -12.0 + 24.0 * (double)i / 200000.0;
    double pdf = exp(-0.5 * z * z) * 0.3989422804014327;  // 1/sqrt(2*pi)
    double sil = z / (1.0 + exp(-z));
    part += sil * sil * pdf;
  }
  red[tid] = part;
  __syncthreads();
  for (int s = 128; s > 0; s >>= 1) {
    if (tid < s) red[tid] += red[tid + s];
    __syncthreads();
  }
  if (tid == 0) {
    double dz = 24.0 / 200000.0;
    cst[0] = (float)(1.0 / sqrt(red[0] * dz));
  }

  if (tid < NPATH) {
    const int L1[NPATH] = {0,0,0,1,1,1,1,2,2,2,2,3,3};
    const int L2[NPATH] = {0,1,2,0,1,1,2,0,1,2,2,1,2};
    const int L3[NPATH] = {0,1,2,1,0,2,1,2,1,0,2,2,1};
    const int WOFF[NPATH] = {0,1,10,35,44,53,98,143,168,213,238,363,468};
    int l1 = L1[tid], l2 = L2[tid], l3 = L3[tid];
    int n1 = 2 * l1 + 1, n2 = 2 * l2 + 1, n3 = 2 * l3 + 1;

    double C[7][5][5];
    su2_cg(l1, l2, l3, C);
    cdbl Q1[7][7], Q2[7][7], Q3[7][7];
    qmat(l1, Q1); qmat(l2, Q2); qmat(l3, Q3);

    double R[7][5][5];
    double nrm = 0.0;
    for (int j = 0; j < n1; j++)
      for (int l = 0; l < n2; l++)
        for (int m = 0; m < n3; m++) {
          cdbl s = {0.0, 0.0};
          for (int i = 0; i < n1; i++)
            for (int k = 0; k < n2; k++)
              for (int n = 0; n < n3; n++) {
                double c = C[i][k][n];
                if (c == 0.0) continue;
                cdbl t = cmul(Q1[i][j], Q2[k][l]);
                cdbl q3c = {Q3[n][m].re, -Q3[n][m].im};
                t = cmul(t, q3c);
                s.re += t.re * c;
                s.im += t.im * c;
              }
          R[j][l][m] = s.re;
          nrm += s.re * s.re;
        }
    nrm = sqrt(nrm);
    float* dst = w3j + WOFF[tid];
    for (int j = 0; j < n1; j++)
      for (int l = 0; l < n2; l++)
        for (int m = 0; m < n3; m++)
          dst[(j * n2 + l) * n3 + m] = (float)(R[j][l][m] / nrm);
  }
}

// ---------------------------------------------------------------------------
// Pack W1/W2/W3 (pre-scaled by 1/sqrt(64)) into f16 WMMA B-fragments.
// Fragment = 32 lanes x 16 halves, contiguous (32B coalesced load per lane).
// Lane 0-15:  col N = ntile*16+lane,    K = kt*32 + {0..7, 16..23}
// Lane 16-31: col N = ntile*16+lane-16, K = kt*32 + {8..15, 24..31}
// ---------------------------------------------------------------------------
__global__ void build_frags(const float* __restrict__ w1, const float* __restrict__ w2,
                            const float* __restrict__ w3, _Float16* __restrict__ frags) {
  int t = blockIdx.x * blockDim.x + threadIdx.x;
  if (t >= NFRAG * 32) return;
  int frag = t >> 5, lane = t & 31;
  const float* W;
  int N, f;
  if (frag < 8)       { W = w1; N = 64;  f = frag; }
  else if (frag < 16) { W = w2; N = 64;  f = frag - 8; }
  else                { W = w3; N = 624; f = frag - 16; }
  int kt = f & 1, ntile = f >> 1;
  int laneN = lane & 15, hi = (lane >> 4) * 8;
  int n = ntile * 16 + laneN;
  _Float16* dst = frags + (size_t)frag * 512 + (size_t)lane * 16;
#pragma unroll
  for (int s = 0; s < 16; s++) {
    int k = kt * 32 + hi + ((s < 8) ? s : (8 + s));
    dst[s] = (_Float16)(W[(size_t)k * N + n] * 0.125f);
  }
}

__global__ void zero_kernel(float4* __restrict__ o, int n4) {
  int i = blockIdx.x * blockDim.x + threadIdx.x;
  if (i < n4) o[i] = make_float4(0.f, 0.f, 0.f, 0.f);
}

// ---------------------------------------------------------------------------
// Main edge kernel: 1 wave = 16 edges. WMMA MLP + fused tensor product +
// atomic scatter into out[dst].
// ---------------------------------------------------------------------------
__global__ __launch_bounds__(WPB * 32, 1) void edge_kernel(
    const float* __restrict__ f_in, const float* __restrict__ pos,
    const int* __restrict__ esrc, const int* __restrict__ edst,
    const float* __restrict__ w0, const _Float16* __restrict__ frags,
    const float* __restrict__ w3j, const float* __restrict__ cstp,
    float* __restrict__ out, int E, float scale) {
  __shared__ alignas(16) _Float16 hA[WPB][16][80];  // padded row stride (160B)
  __shared__ alignas(16) _Float16 hB[WPB][16][80];
  __shared__ float tmpb[WPB][16][NPATH * 5];
  __shared__ float embb[WPB][16][4];
  __shared__ int   dstb[WPB][16];

  const int lane = threadIdx.x & 31;
  const int wid  = threadIdx.x >> 5;
  const int tile = blockIdx.x * WPB + wid;
  const int tiles = (E + 15) >> 4;
  if (tile >= tiles) return;

  const float CST = cstp[0];

  // ----- Phase A: per-edge geometry, SH, radial embed, path tmp -----
  if (lane < 16) {
    const int OFF1[4] = {0, 1, 4, 9};
    const int OFF2[3] = {0, 1, 4};
    const int L1[NPATH] = {0,0,0,1,1,1,1,2,2,2,2,3,3};
    const int L2[NPATH] = {0,1,2,0,1,1,2,0,1,2,2,1,2};
    const int L3[NPATH] = {0,1,2,1,0,2,1,2,1,0,2,2,1};
    const int WOFF[NPATH] = {0,1,10,35,44,53,98,143,168,213,238,363,468};
    const float PWv[3] = {0.57735026919f, 0.77459666924f, 1.0f};

    int e = tile * 16 + lane;
    int de = -1;
    float x1v[16], sh[9], emb[4];
#pragma unroll
    for (int i = 0; i < 16; i++) x1v[i] = 0.f;
#pragma unroll
    for (int i = 0; i < 9; i++) sh[i] = 0.f;
    emb[0] = emb[1] = emb[2] = emb[3] = 0.f;

    if (e < E) {
      int se = esrc[e];
      de = edst[e];
      float dx = pos[3 * de + 0] - pos[3 * se + 0];
      float dy = pos[3 * de + 1] - pos[3 * se + 1];
      float dz = pos[3 * de + 2] - pos[3 * se + 2];
      float r = sqrtf(dx * dx + dy * dy + dz * dz);
      float inv = 1.0f / r;
      float x = dx * inv, y = dy * inv, z = dz * inv;
      const float s3 = 1.7320508075688772f;
      sh[0] = 1.f; sh[1] = x; sh[2] = y; sh[3] = z;
      sh[4] = s3 * x * z; sh[5] = s3 * x * y;
      sh[6] = y * y - 0.5f * (x * x + z * z);
      sh[7] = s3 * y * z; sh[8] = 0.5f * s3 * (z * z - x * x);
#pragma unroll
      for (int k = 0; k < 4; k++) {
        float d = (r - 1.6f * (float)(k + 1)) * 0.625f;
        emb[k] = fast_exp(-d * d) * (2.0f / 1.12f);
      }
#pragma unroll
      for (int i = 0; i < 16; i++) x1v[i] = f_in[(size_t)se * 16 + i];
    }
    dstb[wid][lane] = de;
#pragma unroll
    for (int k = 0; k < 4; k++) embb[wid][lane][k] = emb[k];

#pragma unroll
    for (int p = 0; p < NPATH; p++) {
      const int l1 = L1[p], l2 = L2[p], l3 = L3[p];
      const int n1 = 2 * l1 + 1, n2 = 2 * l2 + 1, n3 = 2 * l3 + 1;
      const float* W = w3j + WOFF[p];
#pragma unroll
      for (int mm = 0; mm < n3; mm++) {
        float s = 0.f;
#pragma unroll
        for (int i = 0; i < n1; i++) {
          float xi = x1v[OFF1[l1] + i];
#pragma unroll
          for (int j = 0; j < n2; j++)
            s += W[(i * n2 + j) * n3 + mm] * xi * sh[OFF2[l2] + j];
        }
        tmpb[wid][lane][p * 5 + mm] = PWv[l3] * s;
      }
    }
  }

  // ----- Layer 0 (K=4, VALU): h0 = CST*silu(emb @ W0/2) -> hA (f16) -----
  {
    int m = lane & 15;
    int nh = lane >> 4;
    float e0 = embb[wid][m][0], e1 = embb[wid][m][1];
    float e2 = embb[wid][m][2], e3 = embb[wid][m][3];
#pragma unroll
    for (int n32 = 0; n32 < 32; n32++) {
      int n = nh * 32 + n32;
      float a = 0.5f * (e0 * w0[n] + e1 * w0[64 + n] + e2 * w0[128 + n] + e3 * w0[192 + n]);
      float hv = CST * a * fast_sigmoid(a);
      hA[wid][m][n] = (_Float16)hv;
    }
  }

  const int hi8 = (lane >> 4) * 8;
  const int Ml = lane & 15;
  const int Mb = (lane >> 4) * 8;

  // A-fragment: lane 0-15 row M=lane, K={0..7,16..23}; lane 16-31 K offset +8.
  auto loadA = [&](const _Float16(&H)[16][80], int kt) -> v16h {
    const _Float16* row = &H[Ml][0];
    v8h c0 = *(const v8h*)(row + kt * 32 + hi8);
    v8h c1 = *(const v8h*)(row + kt * 32 + 16 + hi8);
    v16h a;
#pragma unroll
    for (int s = 0; s < 8; s++) { a[s] = c0[s]; a[8 + s] = c1[s]; }
    return a;
  };
  auto loadB = [&](int frag) -> v16h {
    return *((const v16h*)frags + (size_t)frag * 32 + lane);
  };

  // ----- Layer 1: WMMA (16x64 @ 64x64) -> hB -----
  {
    v16h a0 = loadA(hA[wid], 0);
    v16h a1 = loadA(hA[wid], 1);
#pragma unroll
    for (int nt = 0; nt < 4; nt++) {
      v8f D = {};
      D = __builtin_amdgcn_wmma_f32_16x16x32_f16(false, a0, false, loadB(nt * 2 + 0), (short)0, D, false, false);
      D = __builtin_amdgcn_wmma_f32_16x16x32_f16(false, a1, false, loadB(nt * 2 + 1), (short)0, D, false, false);
#pragma unroll
      for (int r = 0; r < 8; r++) {
        float v = D[r];
        float hv = CST * v * fast_sigmoid(v);
        hB[wid][Mb + r][nt * 16 + Ml] = (_Float16)hv;
      }
    }
  }

  // ----- Layer 2: WMMA -> hA -----
  {
    v16h a0 = loadA(hB[wid], 0);
    v16h a1 = loadA(hB[wid], 1);
#pragma unroll
    for (int nt = 0; nt < 4; nt++) {
      v8f D = {};
      D = __builtin_amdgcn_wmma_f32_16x16x32_f16(false, a0, false, loadB(8 + nt * 2 + 0), (short)0, D, false, false);
      D = __builtin_amdgcn_wmma_f32_16x16x32_f16(false, a1, false, loadB(8 + nt * 2 + 1), (short)0, D, false, false);
#pragma unroll
      for (int r = 0; r < 8; r++) {
        float v = D[r];
        float hv = CST * v * fast_sigmoid(v);
        hA[wid][Mb + r][nt * 16 + Ml] = (_Float16)hv;
      }
    }
  }

  // ----- Layer 3 fused with tensor product + atomic scatter -----
  {
    v16h a0 = loadA(hA[wid], 0);
    v16h a1 = loadA(hA[wid], 1);
    const int GP[NPATH] = {0, 4, 9, 1, 3, 6, 8, 12, 2, 5, 7, 10, 11};  // grouped by l3
    const int GS[4] = {0, 3, 8, 13};
    const int L3B[3] = {0, 48, 192};
#pragma unroll
    for (int g = 0; g < 3; g++) {       // g == l3
      const int n3 = 2 * g + 1;
#pragma unroll
      for (int nt = 0; nt < 3; nt++) {  // 16-channel chunk within 48
        float acc[8][5];
#pragma unroll
        for (int r = 0; r < 8; r++)
#pragma unroll
          for (int mm = 0; mm < 5; mm++) acc[r][mm] = 0.f;

#pragma unroll
        for (int pi = GS[g]; pi < GS[g + 1]; pi++) {
          const int p = GP[pi];
          v8f D = {};
          D = __builtin_amdgcn_wmma_f32_16x16x32_f16(false, a0, false, loadB(16 + (p * 3 + nt) * 2 + 0), (short)0, D, false, false);
          D = __builtin_amdgcn_wmma_f32_16x16x32_f16(false, a1, false, loadB(16 + (p * 3 + nt) * 2 + 1), (short)0, D, false, false);
#pragma unroll
          for (int r = 0; r < 8; r++) {
            float wc = D[r];
#pragma unroll
            for (int mm = 0; mm < n3; mm++)
              acc[r][mm] += tmpb[wid][Mb + r][p * 5 + mm] * wc;
          }
        }
        const int ch = nt * 16 + Ml;
#pragma unroll
        for (int r = 0; r < 8; r++) {
          int d = dstb[wid][Mb + r];
          if (d >= 0) {
            float* op = out + (size_t)d * 432 + L3B[g] + ch * n3;
#pragma unroll
            for (int mm = 0; mm < n3; mm++)
              (void)__hip_atomic_fetch_add(op + mm, scale * acc[r][mm],
                                           __ATOMIC_RELAXED, __HIP_MEMORY_SCOPE_AGENT);
          }
        }
      }
    }
  }
}

// ---------------------------------------------------------------------------
extern "C" void kernel_launch(void* const* d_in, const int* in_sizes, int n_in,
                              void* d_out, int out_size, void* d_ws, size_t ws_size,
                              hipStream_t stream) {
  const float* f_in = (const float*)d_in[0];
  const float* pos  = (const float*)d_in[1];
  const int*   esrc = (const int*)d_in[2];
  const int*   edst = (const int*)d_in[3];
  const float* w0   = (const float*)d_in[4];
  const float* w1   = (const float*)d_in[5];
  const float* w2   = (const float*)d_in[6];
  const float* w3   = (const float*)d_in[7];

  const int E = in_sizes[2];
  const int BN = out_size / 432;
  const float scale = (float)sqrt((double)BN / (double)E);

  _Float16* frags = (_Float16*)d_ws;
  float* w3jp = (float*)((char*)d_ws + (size_t)NFRAG * 512 * 2);
  float* cstp = w3jp + W3J_FLOATS;

  init_tables<<<1, 256, 0, stream>>>(w3jp, cstp);
  build_frags<<<(NFRAG * 32 + 255) / 256, 256, 0, stream>>>(w1, w2, w3, frags);

  int n4 = out_size / 4;
  zero_kernel<<<(n4 + 255) / 256, 256, 0, stream>>>((float4*)d_out, n4);

  int tiles = (E + 15) / 16;
  int blocks = (tiles + WPB - 1) / WPB;
  edge_kernel<<<blocks, WPB * 32, 0, stream>>>(f_in, pos, esrc, edst, w0, frags,
                                               w3jp, cstp, (float*)d_out, E, scale);
}
